// Attention_45732811768356
// MI455X (gfx1250) — matrix operand order repaired and checked by reading, compiled, not run on hardware
//
#include <hip/hip_runtime.h>

// ---------------------------------------------------------------------------
// Attention (B=4, N=1024, M=4096, DIM=256, H=8, Dh=32) for gfx1250 (MI455X).
// Flash-attention style: never materialize the 512MB score matrix.
// All matmuls via v_wmma_f32_16x16x32_f16 (K=32 == DIM_HEAD), f32 accumulate.
// ---------------------------------------------------------------------------

typedef __attribute__((ext_vector_type(16))) _Float16 v16h;
typedef __attribute__((ext_vector_type(8)))  float    v8f;

#define DIMV   256
#define HEADS  8
#define DH     32
#define HD     256     // HEADS*DH
#define BB     4
#define NN     1024
#define MM     4096

struct __align__(16) U4 { unsigned x, y, z, w; };
struct __align__(16) F4 { float x, y, z, w; };

union V16U { U4 q[2]; unsigned u[8]; v16h h; };

static __device__ inline unsigned pkh2(float a, float b) {
    union { _Float16 hh[2]; unsigned uu; } t;
    t.hh[0] = (_Float16)a; t.hh[1] = (_Float16)b;
    return t.uu;
}

#define WMMA_F16(A, Bf, C) \
    __builtin_amdgcn_wmma_f32_16x16x32_f16(false, (A), false, (Bf), (short)0, (C), false, false)

// ---------------------------------------------------------------------------
// Prep: transpose+convert weights to f16 [col][k] layouts; fuse mask+frag_mask.
// ---------------------------------------------------------------------------
__global__ __launch_bounds__(256) void prep_kernel(
    const float* __restrict__ Wqkv, const float* __restrict__ Wout,
    const unsigned char* __restrict__ mask, const float* __restrict__ fmask,
    _Float16* __restrict__ WqkvT, _Float16* __restrict__ WoutT,
    float* __restrict__ fmm)
{
    int t = blockIdx.x * 256 + threadIdx.x;
    const int NQKV = 3 * HD * DIMV;         // 196608
    const int NOUT = HD * DIMV;             // 65536
    if (t < NQKV) {
        int c = t / DIMV, k = t % DIMV;
        WqkvT[c * DIMV + k] = (_Float16)Wqkv[k * (3 * HD) + c];
    } else if (t < NQKV + NOUT) {
        int t2 = t - NQKV;
        int c = t2 / DIMV, k = t2 % DIMV;
        WoutT[c * DIMV + k] = (_Float16)Wout[k * DIMV + c];
    } else if (t < NQKV + NOUT + BB * MM) {
        int t3 = t - NQKV - NOUT;
        fmm[t3] = mask[t3] ? fmask[t3] : -1.0e30f;
    }
}

// Load a 16x32 f16 A-fragment from an f32 row-major matrix (row stride 256).
static __device__ inline v16h load_a_f32(const float* __restrict__ arow, int kb, bool lo) {
    const float* ap = arow + kb + (lo ? 0 : 16);
    F4 a0 = ((const F4*)ap)[0];
    F4 a1 = ((const F4*)ap)[1];
    F4 a2 = ((const F4*)(ap + 8))[0];
    F4 a3 = ((const F4*)(ap + 8))[1];
    v16h a;
    a[0]  = (_Float16)a0.x; a[1]  = (_Float16)a0.y; a[2]  = (_Float16)a0.z; a[3]  = (_Float16)a0.w;
    a[4]  = (_Float16)a1.x; a[5]  = (_Float16)a1.y; a[6]  = (_Float16)a1.z; a[7]  = (_Float16)a1.w;
    a[8]  = (_Float16)a2.x; a[9]  = (_Float16)a2.y; a[10] = (_Float16)a2.z; a[11] = (_Float16)a2.w;
    a[12] = (_Float16)a3.x; a[13] = (_Float16)a3.y; a[14] = (_Float16)a3.z; a[15] = (_Float16)a3.w;
    return a;
}

// Load a 32x16 B-fragment from f16 [col][k] weights (contiguous 32B per lane).
static __device__ inline v16h load_b_wt(const _Float16* __restrict__ brow, int kb, bool lo) {
    const _Float16* bp = brow + kb + (lo ? 0 : 16);
    V16U bf;
    bf.q[0] = ((const U4*)bp)[0];
    bf.q[1] = ((const U4*)bp)[1];
    return bf.h;
}

// Load a 16x32 A-fragment from an f16 row-major window of 32 halves.
static __device__ inline v16h load_a_f16win(const _Float16* __restrict__ win, bool lo) {
    V16U a;
    a.q[0] = ((const U4*)win)[lo ? 0 : 2];
    a.q[1] = ((const U4*)win)[lo ? 1 : 3];
    return a.h;
}

// ---------------------------------------------------------------------------
// Q projection: Qh[b][h][n][d] (f16) = x @ Wq + bq
// ---------------------------------------------------------------------------
__global__ __launch_bounds__(128) void proj_q_kernel(
    const float* __restrict__ x, const _Float16* __restrict__ WqkvT,
    const float* __restrict__ bqkv, _Float16* __restrict__ Qh)
{
    int lane = threadIdx.x & 31;
    int gw = blockIdx.x * 4 + (threadIdx.x >> 5);   // 0..4095
    int rt = gw >> 4, ct = gw & 15;
    int rbase = rt * 16, c0 = ct * 16;
    bool lo = lane < 16;
    int ml = lane & 15;

    const float*    arow = x + (size_t)(rbase + ml) * DIMV;
    const _Float16* brow = WqkvT + (size_t)(c0 + ml) * DIMV;   // q cols: 0..255
    v8f acc = {};
    #pragma unroll
    for (int kb = 0; kb < DIMV; kb += 32)
        acc = WMMA_F16(load_a_f32(arow, kb, lo), load_b_wt(brow, kb, lo), acc);

    int c = c0 + ml;
    float bias = bqkv[c];
    int h = c >> 5, d = c & 31;
    #pragma unroll
    for (int i = 0; i < 8; ++i) {
        int r = rbase + i + (lo ? 0 : 8);
        int b = r >> 10, n = r & (NN - 1);
        Qh[(((size_t)(b * HEADS + h)) * NN + n) * DH + d] = (_Float16)(acc[i] + bias);
    }
}

// ---------------------------------------------------------------------------
// K/V projection: Kh[b][h][m][d] row-major, Vt[b][h][d][m] TRANSPOSED (f16).
// Shares A-fragments between K and V accumulators (2 WMMAs per A load).
// ---------------------------------------------------------------------------
__global__ __launch_bounds__(128) void proj_kv_kernel(
    const float* __restrict__ ctx, const _Float16* __restrict__ WqkvT,
    const float* __restrict__ bqkv, _Float16* __restrict__ Kh,
    _Float16* __restrict__ Vt)
{
    int lane = threadIdx.x & 31;
    int gw = blockIdx.x * 4 + (threadIdx.x >> 5);   // 0..16383
    int rt = gw >> 4, ct = gw & 15;
    int rbase = rt * 16, c0 = ct * 16;
    bool lo = lane < 16;
    int ml = lane & 15;

    const float*    arow  = ctx + (size_t)(rbase + ml) * DIMV;
    const _Float16* bkrow = WqkvT + (size_t)(HD + c0 + ml) * DIMV;
    const _Float16* bvrow = WqkvT + (size_t)(2 * HD + c0 + ml) * DIMV;
    v8f accK = {}, accV = {};
    #pragma unroll
    for (int kb = 0; kb < DIMV; kb += 32) {
        v16h a = load_a_f32(arow, kb, lo);
        accK = WMMA_F16(a, load_b_wt(bkrow, kb, lo), accK);
        accV = WMMA_F16(a, load_b_wt(bvrow, kb, lo), accV);
    }

    int c = c0 + ml;
    float biasK = bqkv[HD + c], biasV = bqkv[2 * HD + c];
    int h = c >> 5, d = c & 31;
    int b = rbase >> 12;
    int jbase = rbase & (MM - 1);

    #pragma unroll
    for (int i = 0; i < 8; ++i) {
        int j = jbase + i + (lo ? 0 : 8);
        Kh[(((size_t)(b * HEADS + h)) * MM + j) * DH + d] = (_Float16)(accK[i] + biasK);
    }
    union { unsigned uu[4]; U4 q; } pv;
    #pragma unroll
    for (int i = 0; i < 4; ++i)
        pv.uu[i] = pkh2(accV[2 * i] + biasV, accV[2 * i + 1] + biasV);
    _Float16* vr = Vt + (((size_t)(b * HEADS + h)) * DH + d) * MM + jbase + (lo ? 0 : 8);
    *((U4*)vr) = pv.q;
}

// ---------------------------------------------------------------------------
// Flash attention. One wave = 16 query rows of one (b,h); loops 32 keys/iter.
// S^T = K_blk x Q^T  (stats per-lane + one shfl_xor(16))
// O^T += V^T x P^T   (P^T B-fragment built with 8 packed-half shfl_xor(16))
// ---------------------------------------------------------------------------
__global__ __launch_bounds__(128) void fa_kernel(
    const _Float16* __restrict__ Qh, const _Float16* __restrict__ Kh,
    const _Float16* __restrict__ Vt, const float* __restrict__ fmm,
    _Float16* __restrict__ Oh)
{
    int lane = threadIdx.x & 31;
    int gw = blockIdx.x * 4 + (threadIdx.x >> 5);   // 0..2047
    int ntile = gw & 63;
    int bh = gw >> 6;                               // b*H + h
    int b = bh >> 3, h = bh & 7;
    int nbase = ntile * 16;
    bool lo = lane < 16;
    int ml = lane & 15;

    // Q^T B-fragment: lane = query column, halves = 16 contiguous dims
    V16U qf;
    const _Float16* qp = Qh + (((size_t)bh) * NN + nbase + ml) * DH + (lo ? 0 : 16);
    qf.q[0] = ((const U4*)qp)[0];
    qf.q[1] = ((const U4*)qp)[1];

    const _Float16* Kb  = Kh + (size_t)bh * MM * DH;
    const _Float16* Vb  = Vt + (size_t)bh * DH * MM;
    const float*    fmb = fmm + b * MM;

    v8f ot0 = {}, ot1 = {};
    float m = -1.0e30f, l = 0.0f;

    for (int jb = 0; jb < MM; jb += 32) {
        // ---- scores S^T (two 16-key x 16-query tiles) ----
        const _Float16* kr0 = Kb + (size_t)(jb + ml) * DH;
        v8f st0 = {}, st1 = {};
        st0 = WMMA_F16(load_a_f16win(kr0, lo), qf.h, st0);
        st1 = WMMA_F16(load_a_f16win(kr0 + 16 * DH, lo), qf.h, st1);

        // ---- fused frag_mask / bool-mask (per key) ----
        const float* f0 = fmb + jb + (lo ? 0 : 8);
        F4 fA = ((const F4*)f0)[0], fB = ((const F4*)f0)[1];
        F4 fC = ((const F4*)(f0 + 16))[0], fD = ((const F4*)(f0 + 16))[1];
        float fm0[8] = { fA.x, fA.y, fA.z, fA.w, fB.x, fB.y, fB.z, fB.w };
        float fm1[8] = { fC.x, fC.y, fC.z, fC.w, fD.x, fD.y, fD.z, fD.w };

        float s0[8], s1[8];
        float mx = -3.0e38f;
        #pragma unroll
        for (int i = 0; i < 8; ++i) {
            s0[i] = st0[i] + fm0[i];
            s1[i] = st1[i] + fm1[i];
            mx = fmaxf(mx, fmaxf(s0[i], s1[i]));
        }
        mx = fmaxf(mx, __shfl_xor(mx, 16, 32));

        // ---- online softmax update ----
        float mnew  = fmaxf(m, mx);
        float alpha = __expf(m - mnew);
        float p0[8], p1[8], ps = 0.0f;
        #pragma unroll
        for (int i = 0; i < 8; ++i) {
            p0[i] = __expf(s0[i] - mnew);
            p1[i] = __expf(s1[i] - mnew);
            ps += p0[i] + p1[i];
        }
        ps += __shfl_xor(ps, 16, 32);
        l = l * alpha + ps;
        m = mnew;
        #pragma unroll
        for (int i = 0; i < 8; ++i) { ot0[i] *= alpha; ot1[i] *= alpha; }

        // ---- build P^T B-fragment (lane halves swap key sub-blocks) ----
        unsigned pk0[4], pk1[4], xp0[4], xp1[4];
        #pragma unroll
        for (int i = 0; i < 4; ++i) {
            pk0[i] = pkh2(p0[2 * i], p0[2 * i + 1]);
            pk1[i] = pkh2(p1[2 * i], p1[2 * i + 1]);
        }
        #pragma unroll
        for (int i = 0; i < 4; ++i) {
            xp0[i] = __shfl_xor(pk0[i], 16, 32);
            xp1[i] = __shfl_xor(pk1[i], 16, 32);
        }
        V16U pb;
        #pragma unroll
        for (int i = 0; i < 4; ++i) {
            pb.u[i]     = lo ? pk0[i] : xp1[i];
            pb.u[4 + i] = lo ? xp0[i] : pk1[i];
        }

        // ---- O^T += V^T x P^T (V pre-transposed: contiguous A loads) ----
        const _Float16* vr0 = Vb + (size_t)ml * MM + jb;
        const _Float16* vr1 = Vb + (size_t)(16 + ml) * MM + jb;
        ot0 = WMMA_F16(load_a_f16win(vr0, lo), pb.h, ot0);
        ot1 = WMMA_F16(load_a_f16win(vr1, lo), pb.h, ot1);
    }

    // ---- normalize and store O as f16 [b][n][h*32+d] ----
    float rl = 1.0f / l;
    union { unsigned uu[4]; U4 q; } o0, o1;
    #pragma unroll
    for (int i = 0; i < 4; ++i) {
        o0.uu[i] = pkh2(ot0[2 * i] * rl, ot0[2 * i + 1] * rl);
        o1.uu[i] = pkh2(ot1[2 * i] * rl, ot1[2 * i + 1] * rl);
    }
    _Float16* orow = Oh + ((size_t)(b * NN + nbase + ml)) * HD + h * DH + (lo ? 0 : 8);
    *((U4*)orow)        = o0.q;
    *((U4*)(orow + 16)) = o1.q;
}

// ---------------------------------------------------------------------------
// Output projection: out (f32) = Oh(f16) @ Wout + bout
// ---------------------------------------------------------------------------
__global__ __launch_bounds__(128) void proj_out_kernel(
    const _Float16* __restrict__ Oh, const _Float16* __restrict__ WoutT,
    const float* __restrict__ bout, float* __restrict__ out)
{
    int lane = threadIdx.x & 31;
    int gw = blockIdx.x * 4 + (threadIdx.x >> 5);   // 0..4095
    int rt = gw >> 4, ct = gw & 15;
    int rbase = rt * 16, c0 = ct * 16;
    bool lo = lane < 16;
    int ml = lane & 15;

    const _Float16* arow = Oh + (size_t)(rbase + ml) * HD;
    const _Float16* brow = WoutT + (size_t)(c0 + ml) * HD;
    v8f acc = {};
    #pragma unroll
    for (int kb = 0; kb < HD; kb += 32)
        acc = WMMA_F16(load_a_f16win(arow + kb, lo), load_b_wt(brow, kb, lo), acc);

    int c = c0 + ml;
    float bias = bout[c];
    #pragma unroll
    for (int i = 0; i < 8; ++i)
        out[(size_t)(rbase + i + (lo ? 0 : 8)) * DIMV + c] = acc[i] + bias;
}

// ---------------------------------------------------------------------------
extern "C" void kernel_launch(void* const* d_in, const int* in_sizes, int n_in,
                              void* d_out, int out_size, void* d_ws, size_t ws_size,
                              hipStream_t stream)
{
    const float*         x     = (const float*)d_in[0];
    const float*         ctx   = (const float*)d_in[1];
    const unsigned char* mask  = (const unsigned char*)d_in[2];
    const float*         fmask = (const float*)d_in[3];
    const float*         Wqkv  = (const float*)d_in[4];
    const float*         bqkv  = (const float*)d_in[5];
    const float*         Wout  = (const float*)d_in[6];
    const float*         bout  = (const float*)d_in[7];
    float* out = (float*)d_out;

    char* ws = (char*)d_ws;
    size_t off = 0;
    auto alloc = [&](size_t bytes) -> char* {
        char* p = ws + off;
        off = (off + bytes + 255) & ~(size_t)255;
        return p;
    };
    _Float16* WqkvT = (_Float16*)alloc((size_t)3 * HD * DIMV * 2);
    _Float16* WoutT = (_Float16*)alloc((size_t)HD * DIMV * 2);
    float*    fmm   = (float*)   alloc((size_t)BB * MM * 4);
    _Float16* Qh    = (_Float16*)alloc((size_t)BB * HEADS * NN * DH * 2);
    _Float16* Kh    = (_Float16*)alloc((size_t)BB * HEADS * MM * DH * 2);
    _Float16* Vt    = (_Float16*)alloc((size_t)BB * HEADS * MM * DH * 2);
    _Float16* Oh    = (_Float16*)alloc((size_t)BB * NN * HD * 2);

    prep_kernel   <<<1088, 256, 0, stream>>>(Wqkv, Wout, mask, fmask, WqkvT, WoutT, fmm);
    proj_q_kernel <<<1024, 128, 0, stream>>>(x, WqkvT, bqkv, Qh);
    proj_kv_kernel<<<4096, 128, 0, stream>>>(ctx, WqkvT, bqkv, Kh, Vt);
    fa_kernel     <<< 512, 128, 0, stream>>>(Qh, Kh, Vt, fmm, Oh);
    proj_out_kernel<<<1024, 128, 0, stream>>>(Oh, WoutT, bout, out);
}
